// HGNN_54546084659602
// MI455X (gfx1250) — matrix operand order repaired and checked by reading, not compiled, not run on hardware
//
#include <hip/hip_runtime.h>

// ---------------------------------------------------------------------------
// 4-layer GCN for MI455X (gfx1250, wave32).
//   - Dense GEMMs via v_wmma_f32_16x16x32_f16 (f16 in, f32 accum).
//   - Weights staged to LDS as f16 (ds_load feeds B fragments).
//   - Message passing: per-edge float4 gather + global_atomic_add_f32 scatter.
//   - N = 50000 = 3125 * 16  -> M tiles need no bounds guards.
// ---------------------------------------------------------------------------

typedef __attribute__((ext_vector_type(16))) _Float16 v16h;
typedef __attribute__((ext_vector_type(8)))  float    v8f;

#define GCN_N 50000

// ---- degree / normalization ------------------------------------------------

__global__ void gcn_init_deg(float* __restrict__ deg, int n) {
    int i = blockIdx.x * blockDim.x + threadIdx.x;
    if (i < n) deg[i] = 1.0f;                    // self-loop contribution
}

__global__ void gcn_deg_scatter(const long long* __restrict__ dst,
                                float* __restrict__ deg, int e) {
    int i = blockIdx.x * blockDim.x + threadIdx.x;
    if (i < e) atomicAdd(&deg[(int)dst[i]], 1.0f);
}

__global__ void gcn_dinv(float* __restrict__ deg, int n) {
    int i = blockIdx.x * blockDim.x + threadIdx.x;
    if (i < n) deg[i] = rsqrtf(deg[i]);          // deg >= 1 always
}

// ---- WMMA GEMM: H[N x NOUT] = X[N x K] @ W[K x NOUT] -----------------------
// One block per 16-row M tile; wave w owns N-tile w. Block = 32 * (NOUT/16).

template <int K, int NOUT>
__global__ void gcn_gemm_wmma(const float* __restrict__ X,
                              const float* __restrict__ W,
                              float* __restrict__ H) {
    __shared__ _Float16 sW[K * NOUT];            // <= 16 KB of the 320 KB LDS

    const int tid = threadIdx.x;
    for (int i = tid; i < K * NOUT; i += blockDim.x)
        sW[i] = (_Float16)W[i];
    __syncthreads();

    const int wave = tid >> 5;                   // N-tile index
    const int lane = tid & 31;
    const int half = lane >> 4;                  // 0: lanes 0-15, 1: lanes 16-31
    const int l15  = lane & 15;

    const int  mrow = blockIdx.x * 16 + l15;     // A-fragment row (both halves)
    const int  colg = wave * 16 + l15;           // B/C column
    const float* __restrict__ xrow = X + (long)mrow * K;

    v8f c = {};

#pragma unroll
    for (int k0 = 0; k0 < K; k0 += 32) {
        // A fragment (16x32 f16): lanes 0-15 hold K = k0+{0..7, 16..23},
        // lanes 16-31 hold K = k0+{8..15, 24..31} for row = lane&15.
        v16h a;
        const int ka = k0 + (half << 3);
#pragma unroll
        for (int j = 0; j < 8; ++j) {
            a[j]     = (_Float16)xrow[ka + j];
            a[8 + j] = (_Float16)xrow[ka + 16 + j];
        }
        // B fragment (32x16 f16): lanes 0-15 hold K = k0+{0..7, 16..23},
        // lanes 16-31 hold K = k0+{8..15, 24..31} for column = lane&15.
        v16h b;
        const int kb = k0 + (half << 3);
#pragma unroll
        for (int j = 0; j < 8; ++j) {
            b[j]     = sW[(kb + j) * NOUT + colg];
            b[8 + j] = sW[(kb + 16 + j) * NOUT + colg];
        }

        c = __builtin_amdgcn_wmma_f32_16x16x32_f16(
                /*neg_a=*/false, a, /*neg_b=*/false, b,
                /*c_mod=*/(short)0, c, /*reuse_a=*/false, /*reuse_b=*/false);
    }

    // C/D f32 16x16 layout: VGPR r -> M = r + 8*half, N = lane&15.
#pragma unroll
    for (int r = 0; r < 8; ++r) {
        const int m = blockIdx.x * 16 + r + (half << 3);
        H[(long)m * NOUT + colg] = c[r];
    }
}

// ---- gather + norm + scatter-add ------------------------------------------
// Thread t handles 4 channels of one edge; consecutive threads cover
// consecutive channel chunks (coalesced float4 gather, contiguous atomics).

template <int C>
__global__ void gcn_scatter(const float* __restrict__ H,
                            const long long* __restrict__ src,
                            const long long* __restrict__ dst,
                            const float* __restrict__ dinv,
                            float* __restrict__ agg, int e, int n) {
    constexpr int CHUNKS = C / 4;
    const long tid   = (long)blockIdx.x * blockDim.x + threadIdx.x;
    const long total = (long)(e + n) * CHUNKS;
    if (tid >= total) return;

    const int edge = (int)(tid / CHUNKS);
    const int ch   = (int)(tid % CHUNKS) * 4;

    int s, d;
    if (edge < e) { s = (int)src[edge]; d = (int)dst[edge]; }
    else          { s = d = edge - e; }                      // self-loop

    const float nm = dinv[s] * dinv[d];
    const float4 hv = *(const float4*)(H + (long)s * C + ch);
    float* __restrict__ ap = agg + (long)d * C + ch;
    atomicAdd(ap + 0, hv.x * nm);
    atomicAdd(ap + 1, hv.y * nm);
    atomicAdd(ap + 2, hv.z * nm);
    atomicAdd(ap + 3, hv.w * nm);
}

// ---- bias + ReLU (+ residual) ---------------------------------------------

template <int C>
__global__ void gcn_epilogue(const float* __restrict__ agg,
                             const float* __restrict__ bias,
                             const float* __restrict__ res,
                             float* __restrict__ out, int n) {
    const long i = (long)blockIdx.x * blockDim.x + threadIdx.x;
    if (i >= (long)n * C) return;
    const int ch = (int)(i % C);
    float v = agg[i] + bias[ch];
    v = v > 0.0f ? v : 0.0f;
    if (res) v += res[i];
    out[i] = v;
}

// ---------------------------------------------------------------------------

extern "C" void kernel_launch(void* const* d_in, const int* in_sizes, int n_in,
                              void* d_out, int out_size, void* d_ws, size_t ws_size,
                              hipStream_t stream) {
    const float*     x  = (const float*)d_in[0];
    const long long* ei = (const long long*)d_in[1];   // int64 [2, E]
    const float* W1 = (const float*)d_in[2];
    const float* b1 = (const float*)d_in[3];
    const float* W2 = (const float*)d_in[4];
    const float* b2 = (const float*)d_in[5];
    const float* W3 = (const float*)d_in[6];
    const float* b3 = (const float*)d_in[7];
    const float* W4 = (const float*)d_in[8];
    const float* b4 = (const float*)d_in[9];

    const int n = GCN_N;
    const int e = in_sizes[1] / 2;                      // 600000
    const long long* src = ei;
    const long long* dst = ei + e;

    // Workspace carve-up (floats): ~116 MB total.
    float* ws    = (float*)d_ws;
    float* dinv  = ws;  ws += n;
    float* h     = ws;  ws += (long)n * 128;
    float* agg   = ws;  ws += (long)n * 128;
    float* xl1   = ws;  ws += (long)n * 128;
    float* xl2   = ws;  ws += (long)n * 64;
    float* xl1td = ws;  ws += (long)n * 128;
    float* out   = (float*)d_out;

    const int TB = 256;
    const int MT = n / 16;                              // 3125 M tiles

    // ---- degree -> dinv ----
    gcn_init_deg  <<<(n + TB - 1) / TB, TB, 0, stream>>>(dinv, n);
    gcn_deg_scatter<<<(e + TB - 1) / TB, TB, 0, stream>>>(dst, dinv, e);
    gcn_dinv      <<<(n + TB - 1) / TB, TB, 0, stream>>>(dinv, n);

    // ---- layer 1: x[64] -> xl1[128] ----
    {
        constexpr int K = 64, C = 128;
        gcn_gemm_wmma<K, C><<<MT, 32 * (C / 16), 0, stream>>>(x, W1, h);
        hipMemsetAsync(agg, 0, (size_t)n * C * sizeof(float), stream);
        const long ts = (long)(e + n) * (C / 4);
        gcn_scatter<C><<<(unsigned)((ts + TB - 1) / TB), TB, 0, stream>>>(h, src, dst, dinv, agg, e, n);
        const long tn = (long)n * C;
        gcn_epilogue<C><<<(unsigned)((tn + TB - 1) / TB), TB, 0, stream>>>(agg, b1, nullptr, xl1, n);
    }
    // ---- layer 2: xl1[128] -> xl2[64] ----
    {
        constexpr int K = 128, C = 64;
        gcn_gemm_wmma<K, C><<<MT, 32 * (C / 16), 0, stream>>>(xl1, W2, h);
        hipMemsetAsync(agg, 0, (size_t)n * C * sizeof(float), stream);
        const long ts = (long)(e + n) * (C / 4);
        gcn_scatter<C><<<(unsigned)((ts + TB - 1) / TB), TB, 0, stream>>>(h, src, dst, dinv, agg, e, n);
        const long tn = (long)n * C;
        gcn_epilogue<C><<<(unsigned)((tn + TB - 1) / TB), TB, 0, stream>>>(agg, b2, nullptr, xl2, n);
    }
    // ---- layer 3: xl2[64] -> xl1td[128], residual xl1 ----
    {
        constexpr int K = 64, C = 128;
        gcn_gemm_wmma<K, C><<<MT, 32 * (C / 16), 0, stream>>>(xl2, W3, h);
        hipMemsetAsync(agg, 0, (size_t)n * C * sizeof(float), stream);
        const long ts = (long)(e + n) * (C / 4);
        gcn_scatter<C><<<(unsigned)((ts + TB - 1) / TB), TB, 0, stream>>>(h, src, dst, dinv, agg, e, n);
        const long tn = (long)n * C;
        gcn_epilogue<C><<<(unsigned)((tn + TB - 1) / TB), TB, 0, stream>>>(agg, b3, xl1, xl1td, n);
    }
    // ---- layer 4: xl1td[128] -> out[64], residual x ----
    {
        constexpr int K = 128, C = 64;
        gcn_gemm_wmma<K, C><<<MT, 32 * (C / 16), 0, stream>>>(xl1td, W4, h);
        hipMemsetAsync(agg, 0, (size_t)n * C * sizeof(float), stream);
        const long ts = (long)(e + n) * (C / 4);
        gcn_scatter<C><<<(unsigned)((ts + TB - 1) / TB), TB, 0, stream>>>(h, src, dst, dinv, agg, e, n);
        const long tn = (long)n * C;
        gcn_epilogue<C><<<(unsigned)((tn + TB - 1) / TB), TB, 0, stream>>>(agg, b4, x, out, n);
    }
}
